// MixedPredictor_38173669327127
// MI455X (gfx1250) — compile-verified, matrix operand order checked by reading, not run on hardware
//
#include <hip/hip_runtime.h>
#include <hip/hip_bf16.h>

typedef __attribute__((ext_vector_type(16))) _Float16 v16h;
typedef __attribute__((ext_vector_type(8)))  _Float16 v8h;
typedef __attribute__((ext_vector_type(8)))  float    v8f;

#define D 128

// Packed f16 weight fragment pools (in halfs), order in d_ws: W1, W2, W3, Wg1, Whead
#define W1_HALFS (256*256)  // 65536  : W1 [256 x 256], KS=8, NT=16   (LDS)
#define W2_HALFS (256*128)  // 32768  : W2 [256 x 128], KS=8, NT=8    (LDS)
#define W3_HALFS (128*64)   // 8192   : W3 [128 x  64], KS=4, NT=4    (global/L2)
#define WG_HALFS (256*64)   // 16384  : Wg1[256 x  64], KS=8, NT=4    (LDS)
#define WH_HALFS (64*16)    // 1024   : Whead[64 x 16]: col0=W4, col1/2=Wg2, KS=2, NT=1 (global/L2)
#define PACK_TOTAL  (W1_HALFS + W2_HALFS + W3_HALFS + WG_HALFS)              // 122880
#define PACK_TOTAL2 (PACK_TOTAL + WH_HALFS)                                  // 123904

// LDS-resident weights: W1 + W2 + Wg1 = 114688 halfs = 229376 B
#define LDS_W_HALFS (W1_HALFS + W2_HALFS + WG_HALFS)

// Per-wave activation scratch (halfs), two M-tiles (32 edges) per wave:
//   x1 : 2 x 16x256 = 8192  (x2 [2 x 16x128] aliases it; head f32 bounce aliases it too)
//   x3 : 2 x 16x64  = 2048
//   gb : 2 x 16x64  = 2048
#define WAVE_SCRATCH_HALFS 12288
#define WAVES_PER_BLOCK 4
#define SMEM_HALFS (LDS_W_HALFS + WAVES_PER_BLOCK * WAVE_SCRATCH_HALFS) // 163840 -> 327680 B (=320KB)

// ---------------------------------------------------------------------------
// K pattern for V_WMMA_*_16X16X32 16-bit operands (ISA 7.12.2):
// lanes 0-15 : K = {0..7, 16..23}; lanes 16-31 : K = {8..15, 24..31}
// ---------------------------------------------------------------------------
__device__ inline void frag_decode(int idx, int KS, int& nt, int& ks, int& lane, int& kl, int& n) {
  int fragId = idx >> 9;           // 512 halfs per (nt,ks) fragment
  int slot   = idx & 511;
  lane = slot >> 4;
  int p = slot & 15;
  ks = fragId % KS;
  nt = fragId / KS;
  int c0 = (lane >= 16) ? 8 : 0;
  kl = (p < 8) ? (p + c0) : (16 + (p - 8) + c0);
  n = nt * 16 + (lane & 15);
}

__device__ inline void pack_one(const float* __restrict__ W, int N, int KS,
                                _Float16* __restrict__ out, int idx) {
  int nt, ks, lane, kl, n;
  frag_decode(idx, KS, nt, ks, lane, kl, n);
  int k = ks * 32 + kl;
  out[idx] = (_Float16)W[k * N + n];
}

__global__ void pack_weights_kernel(const float* __restrict__ W1, const float* __restrict__ W2,
                                    const float* __restrict__ W3, const float* __restrict__ Wg1,
                                    const float* __restrict__ W4, const float* __restrict__ Wg2,
                                    _Float16* __restrict__ ws) {
  int t = blockIdx.x * blockDim.x + threadIdx.x;
  if (t < W1_HALFS) {
    pack_one(W1, 256, 8, ws, t);
  } else if (t < W1_HALFS + W2_HALFS) {
    pack_one(W2, 128, 8, ws + W1_HALFS, t - W1_HALFS);
  } else if (t < W1_HALFS + W2_HALFS + W3_HALFS) {
    pack_one(W3, 64, 4, ws + W1_HALFS + W2_HALFS, t - (W1_HALFS + W2_HALFS));
  } else if (t < PACK_TOTAL) {
    pack_one(Wg1, 64, 8, ws + W1_HALFS + W2_HALFS + W3_HALFS,
             t - (W1_HALFS + W2_HALFS + W3_HALFS));
  } else if (t < PACK_TOTAL2) {
    // Whead [64 x 16]: col0 = W4, col1 = Wg2[:,0], col2 = Wg2[:,1], cols 3..15 = 0
    int idx = t - PACK_TOTAL;
    int nt, ks, lane, kl, n;
    frag_decode(idx, 2, nt, ks, lane, kl, n);
    int k = ks * 32 + kl;
    float v = 0.f;
    if (n == 0)      v = W4[k];
    else if (n == 1) v = Wg2[k * 2 + 0];
    else if (n == 2) v = Wg2[k * 2 + 1];
    ws[t] = (_Float16)v;
  }
}

// ---------------------------------------------------------------------------
// Fragment helpers (wave32)
// ---------------------------------------------------------------------------
__device__ inline v16h lds_bfrag(const _Float16* base, int nt, int ks, int KS, int lane) {
  const _Float16* p = base + ((((nt * KS + ks) << 5) + lane) << 4);  // 32B/lane, 32B aligned
  return *(const v16h*)p;
}

__device__ inline v16h glb_bfrag(const _Float16* base, int nt, int ks, int KS, int lane) {
  const _Float16* p = base + ((((nt * KS + ks) << 5) + lane) << 4);
  return *(const v16h*)p;
}

__device__ inline v16h lds_afrag(const _Float16* x, int ncol, int lane, int ks) {
  int row = lane & 15;
  int c0 = (lane >= 16) ? 8 : 0;
  const _Float16* p0 = x + row * ncol + ks * 32 + c0;   // 16B aligned
  v8h lo = *(const v8h*)p0;
  v8h hi = *(const v8h*)(p0 + 16);
  v16h a;
#pragma unroll
  for (int i = 0; i < 8; i++) { a[i] = lo[i]; a[8 + i] = hi[i]; }
  return a;
}

__device__ inline v16h gather_afrag(const float* __restrict__ srow,
                                    const float* __restrict__ drow, int ks, int lane) {
  const float* row = (ks < 4) ? srow : drow;
  int base = (ks & 3) * 32 + ((lane >= 16) ? 8 : 0);
  float4 f0 = *(const float4*)(row + base);
  float4 f1 = *(const float4*)(row + base + 4);
  float4 f2 = *(const float4*)(row + base + 16);
  float4 f3 = *(const float4*)(row + base + 20);
  v16h a;
  a[0]=(_Float16)f0.x;  a[1]=(_Float16)f0.y;  a[2]=(_Float16)f0.z;  a[3]=(_Float16)f0.w;
  a[4]=(_Float16)f1.x;  a[5]=(_Float16)f1.y;  a[6]=(_Float16)f1.z;  a[7]=(_Float16)f1.w;
  a[8]=(_Float16)f2.x;  a[9]=(_Float16)f2.y;  a[10]=(_Float16)f2.z; a[11]=(_Float16)f2.w;
  a[12]=(_Float16)f3.x; a[13]=(_Float16)f3.y; a[14]=(_Float16)f3.z; a[15]=(_Float16)f3.w;
  return a;
}

__device__ inline void relu_bias(const v8f& c, float bias, v8f& r) {
#pragma unroll
  for (int i = 0; i < 8; i++) { float v = c[i] + bias; r[i] = v > 0.f ? v : 0.f; }
}

__device__ inline void store_cfrag(_Float16* x, int ncol, int lane, int nt, const v8f& c) {
  int col = nt * 16 + (lane & 15);
  int rbase = (lane >= 16) ? 8 : 0;
#pragma unroll
  for (int r = 0; r < 8; r++) x[(r + rbase) * ncol + col] = (_Float16)c[r];
}

#define WMMA_F16(a, b, c) \
  __builtin_amdgcn_wmma_f32_16x16x32_f16(false, (a), false, (b), (short)0, (c), false, false)

// ---------------------------------------------------------------------------
// Fused edge scorer: 32 edges (two M=16 tiles) per wave iteration.
// All GEMM layers share each B fragment across both M-tiles (2 WMMAs per
// fragment fetch); VGPR pressure kept under 256 (no s_set_vgpr_msb).
// ---------------------------------------------------------------------------
__attribute__((amdgpu_waves_per_eu(4)))
__global__ __launch_bounds__(128, 1)
void edge_score_kernel(const float* __restrict__ hU, const float* __restrict__ hI,
                       const int* __restrict__ src, const int* __restrict__ dst,
                       const _Float16* __restrict__ wpack,
                       const float* __restrict__ b1, const float* __restrict__ b2,
                       const float* __restrict__ b3,
                       const float* __restrict__ b4, const float* __restrict__ bg1,
                       const float* __restrict__ bg2,
                       float* __restrict__ out, int nEdges, int nPairs, int totalWaves)
{
  extern __shared__ _Float16 smem[];

  // Stage W1+W2 (contiguous) and Wg1 into LDS; W3/Whead stream from global (L2-hot).
  {
    uint4* d4 = (uint4*)smem;
    const uint4* s4 = (const uint4*)wpack;
    const int nvec1 = (W1_HALFS + W2_HALFS) / 8;
    for (int i = threadIdx.x; i < nvec1; i += blockDim.x) d4[i] = s4[i];
    uint4* d4g = (uint4*)(smem + W1_HALFS + W2_HALFS);
    const uint4* s4g = (const uint4*)(wpack + W1_HALFS + W2_HALFS + W3_HALFS);
    const int nvec2 = WG_HALFS / 8;
    for (int i = threadIdx.x; i < nvec2; i += blockDim.x) d4g[i] = s4g[i];
  }
  __syncthreads();

  const _Float16* w1 = smem;
  const _Float16* w2 = w1 + W1_HALFS;
  const _Float16* wg = w2 + W2_HALFS;
  const _Float16* w3g = wpack + W1_HALFS + W2_HALFS;   // global
  const _Float16* whg = wpack + PACK_TOTAL;            // global head weights

  const int wave = threadIdx.x >> 5;
  const int lane = threadIdx.x & 31;
  const int m    = lane & 15;

  _Float16* scr = smem + LDS_W_HALFS + wave * WAVE_SCRATCH_HALFS;
  _Float16* x1 = scr;             // 2 x 16x256 ; x2 (2 x 16x128) aliases [0,4096)
  _Float16* x3 = scr + 8192;      // 2 x 16x64
  _Float16* gb = scr + 10240;     // 2 x 16x64

  int gwave = blockIdx.x * (blockDim.x >> 5) + wave;

  for (int pair = gwave; pair < nPairs; pair += totalWaves) {
    int base = pair * 32;
    int eA = base + m;          // tile-0 row edge for this lane
    int eB = base + 16 + m;     // tile-1 row edge
    int eAc = eA < nEdges ? eA : 0;
    int eBc = eB < nEdges ? eB : 0;
    int si0 = src[eAc], di0 = dst[eAc];
    int si1 = src[eBc], di1 = dst[eBc];

    int nextE = base + totalWaves * 32;
    if (nextE < nEdges) {
      __builtin_prefetch(src + nextE, 0, 1);
      __builtin_prefetch(dst + nextE, 0, 1);
    }

    const float* s0r = hU + (size_t)si0 * D;
    const float* d0r = hI + (size_t)di0 * D;
    const float* s1r = hU + (size_t)si1 * D;
    const float* d1r = hI + (size_t)di1 * D;

    // A fragments for both tiles: e = concat(s, d), K=256 -> 8 k-steps each
    v16h a0[8], a1v[8];
#pragma unroll
    for (int ks = 0; ks < 8; ks++) {
      a0[ks]  = gather_afrag(s0r, d0r, ks, lane);
      a1v[ks] = gather_afrag(s1r, d1r, ks, lane);
    }

    // Layer 1: [32x256] @ [256x256]
#pragma unroll
    for (int nt = 0; nt < 16; nt++) {
      v8f c0 = {}, c1 = {};
#pragma unroll
      for (int kc = 0; kc < 4; kc++) {
        v16h b0  = lds_bfrag(w1, nt, 2 * kc,     8, lane);
        v16h b1f = lds_bfrag(w1, nt, 2 * kc + 1, 8, lane);
        c0 = WMMA_F16(a0[2 * kc],      b0,  c0);
        c1 = WMMA_F16(a1v[2 * kc],     b0,  c1);
        c0 = WMMA_F16(a0[2 * kc + 1],  b1f, c0);
        c1 = WMMA_F16(a1v[2 * kc + 1], b1f, c1);
      }
      float bias = b1[nt * 16 + m];
      v8f r0, r1; relu_bias(c0, bias, r0); relu_bias(c1, bias, r1);
      store_cfrag(x1,        256, lane, nt, r0);
      store_cfrag(x1 + 4096, 256, lane, nt, r1);
    }

    // Gate hidden: [32x256] @ [256x64]
#pragma unroll
    for (int nt = 0; nt < 4; nt++) {
      v8f c0 = {}, c1 = {};
#pragma unroll
      for (int kc = 0; kc < 4; kc++) {
        v16h b0  = lds_bfrag(wg, nt, 2 * kc,     8, lane);
        v16h b1f = lds_bfrag(wg, nt, 2 * kc + 1, 8, lane);
        c0 = WMMA_F16(a0[2 * kc],      b0,  c0);
        c1 = WMMA_F16(a1v[2 * kc],     b0,  c1);
        c0 = WMMA_F16(a0[2 * kc + 1],  b1f, c0);
        c1 = WMMA_F16(a1v[2 * kc + 1], b1f, c1);
      }
      float bias = bg1[nt * 16 + m];
      v8f r0, r1; relu_bias(c0, bias, r0); relu_bias(c1, bias, r1);
      store_cfrag(gb,        64, lane, nt, r0);
      store_cfrag(gb + 1024, 64, lane, nt, r1);
    }
    __threadfence_block();

    // Layer 2: [32x256] @ [256x128], tiles paired (shared B); x2 aliases x1
    _Float16* x2 = x1;
    {
      v16h A20[8], A21[8];
#pragma unroll
      for (int ks = 0; ks < 8; ks++) {
        A20[ks] = lds_afrag(x1,        256, lane, ks);
        A21[ks] = lds_afrag(x1 + 4096, 256, lane, ks);
      }
      __threadfence_block();  // all reads of x1 precede aliasing stores below
#pragma unroll
      for (int nt = 0; nt < 8; nt++) {
        v8f c0 = {}, c1 = {};
#pragma unroll
        for (int kc = 0; kc < 4; kc++) {
          v16h b0  = lds_bfrag(w2, nt, 2 * kc,     8, lane);
          v16h b1f = lds_bfrag(w2, nt, 2 * kc + 1, 8, lane);
          c0 = WMMA_F16(A20[2 * kc],     b0,  c0);
          c1 = WMMA_F16(A21[2 * kc],     b0,  c1);
          c0 = WMMA_F16(A20[2 * kc + 1], b1f, c0);
          c1 = WMMA_F16(A21[2 * kc + 1], b1f, c1);
        }
        float bias = b2[nt * 16 + m];
        v8f r0, r1; relu_bias(c0, bias, r0); relu_bias(c1, bias, r1);
        store_cfrag(x2,        128, lane, nt, r0);
        store_cfrag(x2 + 2048, 128, lane, nt, r1);
      }
    }
    __threadfence_block();

    // Layer 3: [32x128] @ [128x64], tiles paired, W3 streamed from global
    {
      v16h A30[4], A31[4];
#pragma unroll
      for (int ks = 0; ks < 4; ks++) {
        A30[ks] = lds_afrag(x2,        128, lane, ks);
        A31[ks] = lds_afrag(x2 + 2048, 128, lane, ks);
      }
#pragma unroll
      for (int nt = 0; nt < 4; nt++) {
        v8f c0 = {}, c1 = {};
#pragma unroll
        for (int kc = 0; kc < 2; kc++) {
          v16h b0  = glb_bfrag(w3g, nt, 2 * kc,     4, lane);
          v16h b1f = glb_bfrag(w3g, nt, 2 * kc + 1, 4, lane);
          c0 = WMMA_F16(A30[2 * kc],     b0,  c0);
          c1 = WMMA_F16(A31[2 * kc],     b0,  c1);
          c0 = WMMA_F16(A30[2 * kc + 1], b1f, c0);
          c1 = WMMA_F16(A31[2 * kc + 1], b1f, c1);
        }
        float bias = b3[nt * 16 + m];
        v8f r0, r1; relu_bias(c0, bias, r0); relu_bias(c1, bias, r1);
        store_cfrag(x3,        64, lane, nt, r0);
        store_cfrag(x3 + 1024, 64, lane, nt, r1);
      }
    }
    __threadfence_block();

    // Head GEMMs: [16x64]@[64x16] with Whead (col0=W4, col1/2=Wg2), per tile.
    v16h bh0 = glb_bfrag(whg, 0, 0, 2, lane);
    v16h bh1 = glb_bfrag(whg, 0, 1, 2, lane);
    v8f cm0 = {}, cm1 = {}, cz0 = {}, cz1 = {};
    {
      v16h hx = lds_afrag(x3, 64, lane, 0);
      cm0 = WMMA_F16(hx, bh0, cm0);
      hx = lds_afrag(x3, 64, lane, 1);
      cm0 = WMMA_F16(hx, bh1, cm0);
      hx = lds_afrag(x3 + 1024, 64, lane, 0);
      cm1 = WMMA_F16(hx, bh0, cm1);
      hx = lds_afrag(x3 + 1024, 64, lane, 1);
      cm1 = WMMA_F16(hx, bh1, cm1);
      hx = lds_afrag(gb, 64, lane, 0);
      cz0 = WMMA_F16(hx, bh0, cz0);
      hx = lds_afrag(gb, 64, lane, 1);
      cz0 = WMMA_F16(hx, bh1, cz0);
      hx = lds_afrag(gb + 1024, 64, lane, 0);
      cz1 = WMMA_F16(hx, bh0, cz1);
      hx = lds_afrag(gb + 1024, 64, lane, 1);
      cz1 = WMMA_F16(hx, bh1, cz1);
    }

    // Extract head scalars via small f32 bounce buffer (x1 region is dead).
    // Layout: hb[t*48 + n*16 + m], n=0 -> mlp, n=1 -> z0, n=2 -> z1.
    float* hb = (float*)scr;
    {
      int nsel = m;
      if (nsel < 3) {
        int mb = (lane < 16) ? 0 : 8;
#pragma unroll
        for (int r = 0; r < 8; r++) {
          float v0 = (nsel == 0) ? cm0[r] : cz0[r];
          float v1 = (nsel == 0) ? cm1[r] : cz1[r];
          hb[nsel * 16 + mb + r]      = v0;
          hb[48 + nsel * 16 + mb + r] = v1;
        }
      }
    }
    __threadfence_block();

    // fp32 dot score: lane<16 owns tile-0 edge, lane>=16 owns tile-1 edge
    const float* dsR = (lane < 16) ? s0r : s1r;
    const float* ddR = (lane < 16) ? d0r : d1r;
    float dot = 0.f;
#pragma unroll 4
    for (int k = 0; k < D; k += 4) {
      float4 a4 = *(const float4*)(dsR + k);
      float4 c4 = *(const float4*)(ddR + k);
      dot += a4.x * c4.x + a4.y * c4.y + a4.z * c4.z + a4.w * c4.w;
    }

    {
      int t = lane >> 4;
      int eH = (lane < 16) ? eA : eB;
      if (eH < nEdges) {
        float mlp = hb[t * 48 + m]      + b4[0];
        float z0  = hb[t * 48 + 16 + m] + bg2[0];
        float z1  = hb[t * 48 + 32 + m] + bg2[1];
        float mx  = fmaxf(z0, z1);
        float e0  = __expf(z0 - mx);
        float e1  = __expf(z1 - mx);
        float inv = 1.f / (e0 + e1);
        out[eH] = (e0 * inv) * dot + (e1 * inv) * mlp;
      }
    }
    __threadfence_block();  // hb reads precede next iteration's x1 stores
  }
}

extern "C" void kernel_launch(void* const* d_in, const int* in_sizes, int n_in,
                              void* d_out, int out_size, void* d_ws, size_t ws_size,
                              hipStream_t stream) {
  const float* hU  = (const float*)d_in[0];
  const float* hI  = (const float*)d_in[1];
  const int*   src = (const int*)d_in[2];
  const int*   dst = (const int*)d_in[3];
  const float* W1  = (const float*)d_in[4];
  const float* b1  = (const float*)d_in[5];
  const float* W2  = (const float*)d_in[6];
  const float* b2  = (const float*)d_in[7];
  const float* W3  = (const float*)d_in[8];
  const float* b3  = (const float*)d_in[9];
  const float* W4  = (const float*)d_in[10];
  const float* b4  = (const float*)d_in[11];
  const float* Wg1 = (const float*)d_in[12];
  const float* bg1 = (const float*)d_in[13];
  const float* Wg2 = (const float*)d_in[14];
  const float* bg2 = (const float*)d_in[15];
  float* out = (float*)d_out;

  int nEdges = in_sizes[2];
  _Float16* wpack = (_Float16*)d_ws;

  // 1) repack weights to f16 WMMA fragment order in d_ws
  int packBlocks = (PACK_TOTAL2 + 255) / 256;
  pack_weights_kernel<<<packBlocks, 256, 0, stream>>>(W1, W2, W3, Wg1, W4, Wg2, wpack);

  // 2) fused scorer
  size_t shmem = (size_t)SMEM_HALFS * sizeof(_Float16);  // 327680 B = 320KB WGP LDS
  hipFuncSetAttribute((const void*)edge_score_kernel,
                      hipFuncAttributeMaxDynamicSharedMemorySize, (int)shmem);
  int nPairs = (nEdges + 31) / 32;
  int blocks = 1024;
  int totalWaves = blocks * WAVES_PER_BLOCK;
  edge_score_kernel<<<blocks, 128, shmem, stream>>>(
      hU, hI, src, dst, wpack, b1, b2, b3, b4, bg1, bg2,
      out, nEdges, nPairs, totalWaves);
}
